// SiameseKPConv_2972117369007
// MI455X (gfx1250) — matrix operand order repaired and checked by reading, not compiled, not run on hardware
//
#include <hip/hip_runtime.h>

// ---------------------------------------------------------------------------
// Siamese KPConv forward for MI455X (gfx1250, wave32, WMMA).
// Heavy path: fused gather + bf16 WMMA GEMM (v_wmma_f32_16x16x32_bf16),
// f32 accumulate, 32-row stripes, B staged in LDS. Working set fits in
// 192MB L2 -> compute bound, so we maximize WMMA density per byte moved.
// ---------------------------------------------------------------------------

#define NPTS   32768
#define BATCH  16
#define PPB    2048
#define KNNK   16
#define NKPT   15
#define PIF    0.025f
#define INFL_PITCH (NKPT * KNNK)   // 240 floats per point
#define SLOPE  0.2f
#define EPSBN  1e-5f

typedef __attribute__((ext_vector_type(16))) __bf16 v16bf;
typedef __attribute__((ext_vector_type(8)))  float  v8f;

static __device__ __forceinline__ __bf16 f2bf(float f) {
  union { float f; unsigned u; } a; a.f = f;
  unsigned r = a.u + 0x7FFFu + ((a.u >> 16) & 1u);   // round-to-nearest-even
  unsigned short h = (unsigned short)(r >> 16);
  union { unsigned short s; __bf16 b; } c; c.s = h; return c.b;
}

// ---------------------------------------------------------------------------
// KNN (self, k=16, squared distances) — one thread per query point,
// support positions tiled through LDS. Writes GLOBAL point indices.
// ---------------------------------------------------------------------------
#define KTILE 256
__global__ void knn_kernel(const float* __restrict__ pos, int* __restrict__ idx) {
  const int b = blockIdx.y;
  const int q = b * PPB + blockIdx.x * blockDim.x + threadIdx.x;
  const float qx = pos[q * 3 + 0], qy = pos[q * 3 + 1], qz = pos[q * 3 + 2];

  float bd[KNNK]; int bi[KNNK];
#pragma unroll
  for (int i = 0; i < KNNK; ++i) { bd[i] = 3.0e38f; bi[i] = b * PPB; }

  __shared__ float sp[KTILE * 3];
  for (int s0 = 0; s0 < PPB; s0 += KTILE) {
    __syncthreads();
    for (int t = threadIdx.x; t < KTILE * 3; t += blockDim.x)
      sp[t] = pos[(b * PPB + s0) * 3 + t];
    __syncthreads();
    for (int j = 0; j < KTILE; ++j) {
      const float dx = qx - sp[j * 3 + 0];
      const float dy = qy - sp[j * 3 + 1];
      const float dz = qz - sp[j * 3 + 2];
      const float d = dx * dx + dy * dy + dz * dz;
      if (d < bd[KNNK - 1]) {
        bd[KNNK - 1] = d; bi[KNNK - 1] = b * PPB + s0 + j;
#pragma unroll
        for (int m = KNNK - 2; m >= 0; --m) {
          if (bd[m + 1] < bd[m]) {
            float td = bd[m]; bd[m] = bd[m + 1]; bd[m + 1] = td;
            int   ti = bi[m]; bi[m] = bi[m + 1]; bi[m + 1] = ti;
          }
        }
      }
    }
  }
#pragma unroll
  for (int i = 0; i < KNNK; ++i) idx[q * KNNK + i] = bi[i];
}

// ---------------------------------------------------------------------------
// Cross nearest neighbor: for each point of pq find argmin over ps (same batch)
// ---------------------------------------------------------------------------
__global__ void nn1_kernel(const float* __restrict__ ps, const float* __restrict__ pq,
                           int* __restrict__ cross) {
  const int b = blockIdx.y;
  const int q = b * PPB + blockIdx.x * blockDim.x + threadIdx.x;
  const float qx = pq[q * 3 + 0], qy = pq[q * 3 + 1], qz = pq[q * 3 + 2];
  float best = 3.0e38f; int bi = b * PPB;

  __shared__ float sp[KTILE * 3];
  for (int s0 = 0; s0 < PPB; s0 += KTILE) {
    __syncthreads();
    for (int t = threadIdx.x; t < KTILE * 3; t += blockDim.x)
      sp[t] = ps[(b * PPB + s0) * 3 + t];
    __syncthreads();
    for (int j = 0; j < KTILE; ++j) {
      const float dx = qx - sp[j * 3 + 0];
      const float dy = qy - sp[j * 3 + 1];
      const float dz = qz - sp[j * 3 + 2];
      const float d = dx * dx + dy * dy + dz * dz;
      if (d < best) { best = d; bi = b * PPB + s0 + j; }
    }
  }
  cross[q] = bi;
}

// ---------------------------------------------------------------------------
// Kernel-point influences: infl[n][p*16+j] = max(0, 1 - |rel(n,j) - kp_p|/PI)
// Depends only on pos+idx -> computed once per cloud, reused by every layer.
// ---------------------------------------------------------------------------
__global__ void infl_kernel(const float* __restrict__ pos, const int* __restrict__ idx,
                            const float* __restrict__ kp, float* __restrict__ infl) {
  const int t = blockIdx.x * blockDim.x + threadIdx.x;   // over NPTS*KNNK
  if (t >= NPTS * KNNK) return;
  const int n = t >> 4, j = t & 15;
  const int nb = idx[n * KNNK + j];
  const float rx = pos[nb * 3 + 0] - pos[n * 3 + 0];
  const float ry = pos[nb * 3 + 1] - pos[n * 3 + 1];
  const float rz = pos[nb * 3 + 2] - pos[n * 3 + 2];
  for (int p = 0; p < NKPT; ++p) {
    const float dx = rx - kp[p * 3 + 0];
    const float dy = ry - kp[p * 3 + 1];
    const float dz = rz - kp[p * 3 + 2];
    const float dist = sqrtf(dx * dx + dy * dy + dz * dz);
    infl[n * INFL_PITCH + p * KNNK + j] = fmaxf(0.0f, 1.0f - dist / PIF);
  }
}

// ---------------------------------------------------------------------------
// Fused KPConv: 32-row output stripe x all Cout per workgroup.
// Block = (Cout/16) waves of 32. Per (kernel-point p, 32-wide K chunk):
//   1) cooperative influence-weighted gather -> 32x32 bf16 A-tile in LDS
//   2) cooperative coalesced stage of W chunk -> 32xCout bf16 B in LDS
//   3) each wave: one B fragment, two A fragments, two
//      v_wmma_f32_16x16x32_bf16 with persistent f32 accumulators.
// ---------------------------------------------------------------------------
__global__ void __launch_bounds__(512)
kpconv_wmma_kernel(const float* __restrict__ xA, int CA,
                   const float* __restrict__ xB, int CB,
                   const int* __restrict__ idx,
                   const float* __restrict__ infl,
                   const float* __restrict__ W,       // [NKPT][Cin][Cout] f32
                   float* __restrict__ out, int Cout) {
  const int Cin  = CA + CB;
  const int row0 = blockIdx.x * 32;
  const int T    = blockDim.x;
  const int tid  = threadIdx.x;
  const int lane = tid & 31;
  const int wv   = tid >> 5;            // column tile owned by this wave

  __shared__ int    sIdx[512];          // 32 rows x 16 neighbors
  __shared__ float  sInfl[512];         // 32 rows x 16 neighbors (current p)
  __shared__ __bf16 sA[1024];           // 32 rows x 32 K-slice, bf16
  extern __shared__ __bf16 sB[];        // 32 K x Cout, bf16 (dynamic)

  for (int t = tid; t < 512; t += T) {
    const int r = t >> 4, j = t & 15;
    sIdx[t] = idx[(row0 + r) * KNNK + j];
  }

  v8f acc0, acc1;
#pragma unroll
  for (int i = 0; i < 8; ++i) { acc0[i] = 0.0f; acc1[i] = 0.0f; }

  for (int p = 0; p < NKPT; ++p) {
    __builtin_prefetch(W + (size_t)(p * Cin) * Cout, 0, 1);  // global_prefetch_b8
    __syncthreads();
    for (int t = tid; t < 512; t += T) {
      const int r = t >> 4, j = t & 15;
      sInfl[t] = infl[(row0 + r) * INFL_PITCH + p * KNNK + j];
    }
    for (int c0 = 0; c0 < Cin; c0 += 32) {
      __syncthreads();
      // 1) influence-weighted gather into the 32x32 bf16 A-tile
      for (int t = tid; t < 1024; t += T) {
        const int r = t >> 5, kk = t & 31;
        const int c = c0 + kk;
        float s = 0.0f;
        if (c < Cin) {
          const float* src = (c < CA) ? xA : xB;
          const int cc     = (c < CA) ? c : (c - CA);
          const int pitch  = (c < CA) ? CA : CB;
#pragma unroll
          for (int j = 0; j < KNNK; ++j)
            s += sInfl[r * 16 + j] * src[sIdx[r * 16 + j] * pitch + cc];
        }
        sA[t] = f2bf(s);
      }
      // 2) coalesced stage of the 32xCout weight chunk, f32 -> bf16
      const int nB = 32 * Cout;
      for (int t = tid; t < nB; t += T) {
        const int kk  = t / Cout;
        const int col = t - kk * Cout;
        const int kg  = c0 + kk;
        sB[t] = f2bf((kg < Cin) ? W[(size_t)(p * Cin + kg) * Cout + col] : 0.0f);
      }
      __syncthreads();

      // 3) fragments + two WMMAs (lane layouts per ISA 7.12.2)
      v16bf a0, a1, b;
      const int rr = lane & 15;
      const int ah = (lane & 16) ? 8 : 0;    // A half offset
      const int bh = (lane & 16) ? 8 : 0;    // B half offset
      const int colg = wv * 16 + rr;
#pragma unroll
      for (int e = 0; e < 16; ++e) {
        const int Ka = e + (e & 8) + ah;     // e + 8*(e>=8) + 8*(lane>=16)
        a0[e] = sA[rr * 32 + Ka];
        a1[e] = sA[(16 + rr) * 32 + Ka];
        b[e]  = sB[(e + (e & 8) + bh) * Cout + colg];  // K = e + 8*(e>=8) + 8*(lane>=16)
      }
      acc0 = __builtin_amdgcn_wmma_f32_16x16x32_bf16(
          false, a0, false, b, (short)0, acc0, false, false);
      acc1 = __builtin_amdgcn_wmma_f32_16x16x32_bf16(
          false, a1, false, b, (short)0, acc1, false, false);
    }
  }

  // D layout: lane -> col = lane&15; VGPR r -> row = r + 8*(lane>=16)
  const int col = wv * 16 + (lane & 15);
  const int rb0 = row0 + ((lane & 16) ? 8 : 0);
  const int rb1 = rb0 + 16;
#pragma unroll
  for (int r = 0; r < 8; ++r) {
    out[(size_t)(rb0 + r) * Cout + col] = acc0[r];
    out[(size_t)(rb1 + r) * Cout + col] = acc1[r];
  }
}

// ---------------------------------------------------------------------------
// Per-channel mean / (biased) variance over NPTS rows. One block per channel.
// ---------------------------------------------------------------------------
__global__ void stats_kernel(const float* __restrict__ x, int C,
                             float* __restrict__ mu, float* __restrict__ var) {
  const int c = blockIdx.x;
  float s = 0.0f, s2 = 0.0f;
  for (int n = threadIdx.x; n < NPTS; n += blockDim.x) {
    const float v = x[(size_t)n * C + c];
    s += v; s2 += v * v;
  }
  __shared__ float rs[256], rq[256];
  rs[threadIdx.x] = s; rq[threadIdx.x] = s2;
  __syncthreads();
  for (int o = 128; o > 0; o >>= 1) {
    if (threadIdx.x < o) { rs[threadIdx.x] += rs[threadIdx.x + o];
                           rq[threadIdx.x] += rq[threadIdx.x + o]; }
    __syncthreads();
  }
  if (threadIdx.x == 0) {
    const float m = rs[0] / (float)NPTS;
    mu[c]  = m;
    var[c] = rq[0] / (float)NPTS - m * m;
  }
}

// ---------------------------------------------------------------------------
// In-place batch-norm + leaky ReLU
// ---------------------------------------------------------------------------
__global__ void bn_leaky_kernel(float* __restrict__ x, int C,
                                const float* __restrict__ mu, const float* __restrict__ var,
                                const float* __restrict__ g,  const float* __restrict__ b) {
  const int t = blockIdx.x * blockDim.x + threadIdx.x;
  if (t >= NPTS * C) return;
  const int c = t % C;
  const float v = g[c] * (x[t] - mu[c]) * rsqrtf(var[c] + EPSBN) + b[c];
  x[t] = (v >= 0.0f) ? v : SLOPE * v;
}

// ---------------------------------------------------------------------------
// diff[n,c] = h2[n,c] - h1[cross[n], c]
// ---------------------------------------------------------------------------
__global__ void diff_kernel(const float* __restrict__ h2, const float* __restrict__ h1,
                            const int* __restrict__ cross, float* __restrict__ out, int C) {
  const int t = blockIdx.x * blockDim.x + threadIdx.x;
  if (t >= NPTS * C) return;
  const int n = t / C, c = t % C;
  out[t] = h2[t] - h1[(size_t)cross[n] * C + c];
}

// ---------------------------------------------------------------------------
// f = h(Nx32) @ fc1_W(32x64)
// ---------------------------------------------------------------------------
__global__ void fc1_kernel(const float* __restrict__ h, const float* __restrict__ Wf,
                           float* __restrict__ f) {
  const int t = blockIdx.x * blockDim.x + threadIdx.x;  // over NPTS*64
  if (t >= NPTS * 64) return;
  const int n = t >> 6, o = t & 63;
  float s = 0.0f;
#pragma unroll
  for (int c = 0; c < 32; ++c) s += h[n * 32 + c] * Wf[c * 64 + o];
  f[t] = s;
}

// ---------------------------------------------------------------------------
// logits = f(Nx64) @ fc2_W(64x2); log_softmax over 2 classes
// ---------------------------------------------------------------------------
__global__ void head_kernel(const float* __restrict__ f, const float* __restrict__ W2,
                            float* __restrict__ out) {
  const int n = blockIdx.x * blockDim.x + threadIdx.x;
  if (n >= NPTS) return;
  float l0 = 0.0f, l1 = 0.0f;
#pragma unroll
  for (int c = 0; c < 64; ++c) {
    const float v = f[n * 64 + c];
    l0 += v * W2[c * 2 + 0];
    l1 += v * W2[c * 2 + 1];
  }
  const float m   = fmaxf(l0, l1);
  const float lse = m + logf(expf(l0 - m) + expf(l1 - m));
  out[n * 2 + 0] = l0 - lse;
  out[n * 2 + 1] = l1 - lse;
}

// ---------------------------------------------------------------------------
// Host orchestration
// ---------------------------------------------------------------------------
extern "C" void kernel_launch(void* const* d_in, const int* in_sizes, int n_in,
                              void* d_out, int out_size, void* d_ws, size_t ws_size,
                              hipStream_t stream) {
  (void)in_sizes; (void)n_in; (void)out_size; (void)ws_size;

  const float* x1   = (const float*)d_in[0];
  const float* pos1 = (const float*)d_in[1];
  const float* x2   = (const float*)d_in[2];
  const float* pos2 = (const float*)d_in[3];
  const float* kp   = (const float*)d_in[4];
  const float* dW[4] = {(const float*)d_in[5],  (const float*)d_in[6],
                        (const float*)d_in[7],  (const float*)d_in[8]};
  const float* dg[4] = {(const float*)d_in[9],  (const float*)d_in[10],
                        (const float*)d_in[11], (const float*)d_in[12]};
  const float* db[4] = {(const float*)d_in[13], (const float*)d_in[14],
                        (const float*)d_in[15], (const float*)d_in[16]};
  const float* uW[3] = {(const float*)d_in[17], (const float*)d_in[18],
                        (const float*)d_in[19]};
  const float* ug[3] = {(const float*)d_in[20], (const float*)d_in[21],
                        (const float*)d_in[22]};
  const float* ub[3] = {(const float*)d_in[23], (const float*)d_in[24],
                        (const float*)d_in[25]};
  const float* fc1W = (const float*)d_in[26];
  const float* fcg  = (const float*)d_in[27];
  const float* fcb  = (const float*)d_in[28];
  const float* fc2W = (const float*)d_in[29];

  // ---- workspace layout (~230 MB of f32) ----
  float* w = (float*)d_ws;
  size_t o = 0;
  int*   idx1  = (int*)(w + o); o += (size_t)NPTS * KNNK;
  int*   idx2  = (int*)(w + o); o += (size_t)NPTS * KNNK;
  int*   cross = (int*)(w + o); o += NPTS;
  float* infl1 = w + o; o += (size_t)NPTS * INFL_PITCH;
  float* infl2 = w + o; o += (size_t)NPTS * INFL_PITCH;
  float* P1 = w + o; o += (size_t)NPTS * 256;
  float* P2 = w + o; o += (size_t)NPTS * 256;
  float* P3 = w + o; o += (size_t)NPTS * 256;
  float* P4 = w + o; o += (size_t)NPTS * 256;
  float* D0 = w + o; o += (size_t)NPTS * 32;
  float* D1 = w + o; o += (size_t)NPTS * 64;
  float* D2 = w + o; o += (size_t)NPTS * 128;
  float* mu  = w + o; o += 256;
  float* var = w + o; o += 256;

  // ---- geometry (positions fixed for the whole forward pass) ----
  dim3 pgrid(PPB / 256, BATCH);
  knn_kernel<<<pgrid, 256, 0, stream>>>(pos1, idx1);
  knn_kernel<<<pgrid, 256, 0, stream>>>(pos2, idx2);
  nn1_kernel<<<pgrid, 256, 0, stream>>>(pos1, pos2, cross);
  infl_kernel<<<(NPTS * KNNK + 255) / 256, 256, 0, stream>>>(pos1, idx1, kp, infl1);
  infl_kernel<<<(NPTS * KNNK + 255) / 256, 256, 0, stream>>>(pos2, idx2, kp, infl2);

  auto conv_bn = [&](const float* inA, int CA, const float* inB, int CB,
                     const int* idx, const float* infl, const float* W,
                     const float* g, const float* b, float* ob, int Cout) {
    const size_t shmem = (size_t)(32 * Cout) * sizeof(__bf16);   // dynamic B stage
    kpconv_wmma_kernel<<<dim3(NPTS / 32), dim3((Cout / 16) * 32), shmem, stream>>>(
        inA, CA, inB, CB, idx, infl, W, ob, Cout);
    stats_kernel<<<dim3(Cout), dim3(256), 0, stream>>>(ob, Cout, mu, var);
    const int tot = NPTS * Cout;
    bn_leaky_kernel<<<(tot + 255) / 256, 256, 0, stream>>>(ob, Cout, mu, var, g, b);
  };
  auto diff = [&](const float* h2, const float* h1, float* ob, int C) {
    const int tot = NPTS * C;
    diff_kernel<<<(tot + 255) / 256, 256, 0, stream>>>(h2, h1, cross, ob, C);
  };

  // ---- down path (shared weights across the two clouds) ----
  conv_bn(x1, 3, nullptr, 0, idx1, infl1, dW[0], dg[0], db[0], P1, 32);
  conv_bn(x2, 3, nullptr, 0, idx2, infl2, dW[0], dg[0], db[0], P3, 32);
  diff(P3, P1, D0, 32);
  conv_bn(P1, 32, nullptr, 0, idx1, infl1, dW[1], dg[1], db[1], P2, 64);
  conv_bn(P3, 32, nullptr, 0, idx2, infl2, dW[1], dg[1], db[1], P4, 64);
  diff(P4, P2, D1, 64);
  conv_bn(P2, 64, nullptr, 0, idx1, infl1, dW[2], dg[2], db[2], P1, 128);
  conv_bn(P4, 64, nullptr, 0, idx2, infl2, dW[2], dg[2], db[2], P3, 128);
  diff(P3, P1, D2, 128);
  conv_bn(P1, 128, nullptr, 0, idx1, infl1, dW[3], dg[3], db[3], P2, 256);
  conv_bn(P3, 128, nullptr, 0, idx2, infl2, dW[3], dg[3], db[3], P4, 256);
  diff(P4, P2, P1, 256);                       // h (bottom) -> P1

  // ---- up path (cloud-2 geometry), channel concat [h | stack.pop()] ----
  conv_bn(P1, 256, D2, 128, idx2, infl2, uW[0], ug[0], ub[0], P2, 128);
  conv_bn(P2, 128, D1, 64,  idx2, infl2, uW[1], ug[1], ub[1], P3, 64);
  conv_bn(P3, 64,  D0, 32,  idx2, infl2, uW[2], ug[2], ub[2], P4, 32);

  // ---- FC head ----
  fc1_kernel<<<(NPTS * 64 + 255) / 256, 256, 0, stream>>>(P4, fc1W, P1);
  stats_kernel<<<dim3(64), dim3(256), 0, stream>>>(P1, 64, mu, var);
  bn_leaky_kernel<<<(NPTS * 64 + 255) / 256, 256, 0, stream>>>(P1, 64, mu, var, fcg, fcb);
  head_kernel<<<(NPTS + 255) / 256, 256, 0, stream>>>(P1, fc2W, (float*)d_out);
}
